// LA_GMF_83811991814234
// MI455X (gfx1250) — matrix operand order repaired and checked by reading, not compiled, not run on hardware
//
#include <hip/hip_runtime.h>

typedef __attribute__((ext_vector_type(16))) _Float16 v16h;
typedef __attribute__((ext_vector_type(8)))  float    v8f;
typedef _Float16 half_t;

// ---------------------------------------------------------------------------
// Workspace layout (bytes). Conv-chain activations are f16 NDHWC.
// ---------------------------------------------------------------------------
static const size_t OFF_H    = 0;                       // 19,660,800 halves
static const size_t OFF_A    = 39321600;                // 19,660,800 halves
static const size_t OFF_B2   = 78643200;                // out14 (NDHWC) lives here
static const size_t OFF_CAT  = 117964800;               // 39,321,600 halves
static const size_t OFF_W    = 196608000;               // f16 swizzled weights
static const size_t OFF_SB   = OFF_W + 942080;          // 11 convs * 4096B scale/bias
static const size_t OFF_HEAD = OFF_SB + 45056;
// head region (f32)
static const size_t OFF_TOK = OFF_HEAD;                 // 307200 B
static const size_t OFF_P2  = OFF_TOK + 307200;         // 307200
static const size_t OFF_HH  = OFF_P2  + 307200;         // 76800
static const size_t OFF_AL  = OFF_HH  + 76800;          // 1280
static const size_t OFF_IDX = OFF_AL  + 1280;           // 256 (int)
static const size_t OFF_G1  = OFF_IDX + 256;            // 20480
static const size_t OFF_G2  = OFF_G1  + 20480;
static const size_t OFF_AH1 = OFF_G2  + 20480;          // 1024
static const size_t OFF_AH2 = OFF_AH1 + 1024;
static const size_t OFF_XW  = OFF_AH2 + 1024;           // 20480
static const size_t OFF_GO1 = OFF_XW  + 20480;
static const size_t OFF_GO2 = OFF_GO1 + 20480;
static const size_t OFF_FUS = OFF_GO2 + 20480;          // 40960
static const size_t OFF_Q   = OFF_FUS + 40960;
static const size_t OFF_K   = OFF_Q   + 40960;
static const size_t OFF_V   = OFF_K   + 40960;
static const size_t OFF_GC  = OFF_V   + 40960;          // 40960
static const size_t OFF_HA  = OFF_GC  + 40960;          // 20480
static const size_t OFF_CM  = OFF_HA  + 20480;          // 1024
static const size_t OFF_WN  = OFF_CM  + 1024;           // 1024
static const size_t OFF_GS  = OFF_WN  + 1024;           // 2048

// ---------------------------------------------------------------------------
// Prep: fold BN+bias into per-channel scale/bias; swizzle weights into the
// WMMA A-fragment layout this code uses (A = 16 OC rows x 32 K), tap-major GEMM-K:
//   k = tap*IC + ic ; per lane: m = lane&15 (=oc), k = (lane>>4)*16+h
// wfrag layout: [ocTile][kStep][lane][16 halves]
// ---------------------------------------------------------------------------
__global__ void prep_sb(const float* __restrict__ g, const float* __restrict__ be,
                        const float* __restrict__ mn, const float* __restrict__ vv,
                        const float* __restrict__ cb, float* __restrict__ sb, int OC) {
  int i = threadIdx.x + blockIdx.x * blockDim.x;
  if (i >= OC) return;
  float s = g[i] * rsqrtf(vv[i] + 1e-5f);
  sb[i]      = s;
  sb[OC + i] = cb[i] * s + be[i] - mn[i] * s;
}

__global__ void prep_wfrag(const float* __restrict__ w, half_t* __restrict__ wf,
                           int OC, int IC, int KS3) {
  int K = IC * KS3;
  int Kp = (K + 31) & ~31;
  int tot = OC * Kp;
  int i = threadIdx.x + blockIdx.x * blockDim.x;
  if (i >= tot) return;
  int h    = i & 15;
  int lane = (i >> 4) & 31;
  int rest = i >> 9;
  int nks  = Kp >> 5;
  int kStep = rest % nks;
  int ocT   = rest / nks;
  int n = ocT * 16 + (lane & 15);
  int k = kStep * 32 + ((lane >> 4) << 4) + h;
  half_t v = (half_t)0.f;
  if (k < K) {
    int tap = k / IC;
    int ic  = k - tap * IC;
    v = (half_t)w[((size_t)n * IC + ic) * KS3 + tap];   // OIDHW source
  }
  wf[i] = v;
}

// ---------------------------------------------------------------------------
// Stem: conv k=2 stride=2 (1->16) + BN + relu, fp32 NCDHW in -> f16 NDHWC out
// ---------------------------------------------------------------------------
__global__ void stem_kernel(const float* __restrict__ x, const float* __restrict__ w,
                            const float* __restrict__ sb, half_t* __restrict__ out, int B) {
  const int DHW = 80 * 96 * 80;
  long long i = (long long)blockIdx.x * blockDim.x + threadIdx.x;
  if (i >= (long long)B * DHW * 16) return;
  int oc = (int)(i & 15);
  int m  = (int)((i >> 4) % DHW);
  int b  = (int)(i / ((long long)DHW * 16));
  int z = m / (96 * 80); int r = m % (96 * 80); int y = r / 80; int xx = r % 80;
  const float* xp = x + (size_t)b * 160 * 192 * 160;
  float acc = 0.f;
  #pragma unroll
  for (int dz = 0; dz < 2; dz++)
    #pragma unroll
    for (int dy = 0; dy < 2; dy++)
      #pragma unroll
      for (int dx = 0; dx < 2; dx++)
        acc += xp[(size_t)(2 * z + dz) * 192 * 160 + (2 * y + dy) * 160 + (2 * xx + dx)]
             * w[oc * 8 + dz * 4 + dy * 2 + dx];
  float val = acc * sb[oc] + sb[16 + oc];
  out[i] = (half_t)fmaxf(val, 0.f);   // NDHWC linear
}

// ---------------------------------------------------------------------------
// Implicit-GEMM conv3d (stride 1), NDHWC f16, v_wmma_f32_16x16x32_f16.
// A = weight fragment (1x 32B load/lane/step, L2-resident, pre-swizzled)
// B = activations    (1x 32B contiguous load/lane/step: 16 channels @ 1 tap)
// D rows = OC -> contiguous channel stores.
// mode 0: bn    1: relu(bn)    2: relu(aIn + bn) -> out (+opt raw b2 to out2)
// ---------------------------------------------------------------------------
template <int KS, int IC>
__global__ __launch_bounds__(128)
void conv_wmma_kernel(const half_t* __restrict__ in, const half_t* __restrict__ wfrag,
                      const float* __restrict__ sb, const half_t* __restrict__ aIn,
                      half_t* __restrict__ out, half_t* __restrict__ out2,
                      int OC, int D, int H, int W, int pad, int mode,
                      int outC, int outCOff) {
  const int DHW = D * H * W;
  const int KS3 = KS * KS * KS;
  const int K   = IC * KS3;
  const int Kp  = (K + 31) & ~31;
  const int lane  = threadIdx.x & 31;
  const int wv    = threadIdx.x >> 5;
  const int mBase = (blockIdx.x * 4 + wv) * 16;
  if (mBase >= DHW) return;                      // uniform per wave
  const int ocT = blockIdx.y;
  const int b   = blockIdx.z;

  const int pos = mBase + (lane & 15);           // this lane's spatial position
  const bool mOK = pos < DHW;
  int z0, y0, x0;
  {
    int mm = mOK ? pos : 0;
    z0 = mm / (H * W); int r = mm - z0 * H * W; y0 = r / W; x0 = r - y0 * W;
  }
  const int hi = lane >> 4;
  const half_t* inB = in + (size_t)b * DHW * IC;
  const half_t* wp  = wfrag + (size_t)ocT * (Kp >> 5) * 512 + lane * 16;

  v8f acc = {};
  for (int kStep = 0; kStep < Kp; kStep += 32) {
    // A: weight fragment, one 32B load
    v16h a = *(const v16h*)(wp + (size_t)(kStep >> 5) * 512);
    __builtin_prefetch((const void*)(wp + (size_t)((kStep >> 5) + 1) * 512), 0, 3);
    // B: 16 contiguous input channels at one tap for this lane's position
    int k0 = kStep + (hi << 4);
    int tap = k0 / IC;                           // compile-time shift
    int icBase = k0 - tap * IC;
    int dz, dy, dx;
    if (KS == 3) { dz = tap / 9; int r = tap - dz * 9; dy = r / 3; dx = r - dy * 3; }
    else { dz = dy = dx = 0; }
    int zz = z0 + dz - pad, yy = y0 + dy - pad, xx = x0 + dx - pad;
    bool ok = mOK && (tap < KS3) &&
              (unsigned)zz < (unsigned)D && (unsigned)yy < (unsigned)H &&
              (unsigned)xx < (unsigned)W;
    size_t bidx = ok ? ((size_t)((zz * H + yy) * W + xx) * IC + icBase) : 0;
    v16h bf = *(const v16h*)(inB + bidx);
    v16h zv = {};
    if (!ok) bf = zv;
    acc = __builtin_amdgcn_wmma_f32_16x16x32_f16(false, a, false, bf,
                                                 (short)0, acc, false, false);
  }

  if (!mOK) return;
  const int ocBase = ocT * 16 + (hi << 3);       // 8 consecutive output channels
  size_t obase = ((size_t)b * DHW + pos) * outC + outCOff + ocBase;
  size_t abase = ((size_t)b * DHW + pos) * OC + ocBase;
  #pragma unroll
  for (int r = 0; r < 8; ++r) {
    int oc = ocBase + r;
    float val = acc[r] * sb[oc] + sb[OC + oc];
    if (mode == 0) {
      out[obase + r] = (half_t)val;
    } else if (mode == 1) {
      out[obase + r] = (half_t)fmaxf(val, 0.f);
    } else {
      float av = (float)aIn[abase + r];
      out[obase + r] = (half_t)fmaxf(av + val, 0.f);
      if (out2) out2[abase + r] = (half_t)val;
    }
  }
}

// ---------------------------------------------------------------------------
// Pool / concat-copy / tok1 (NDHWC f16, channel innermost -> coalesced)
// ---------------------------------------------------------------------------
__global__ void pool_kernel(const half_t* __restrict__ in, half_t* __restrict__ out,
                            int B, int C, int D, int H, int W) {
  int oD = D / 2, oH = H / 2, oW = W / 2; int oDHW = oD * oH * oW;
  long long i = (long long)blockIdx.x * blockDim.x + threadIdx.x;
  if (i >= (long long)B * oDHW * C) return;
  int c = (int)(i % C);
  int m = (int)((i / C) % oDHW);
  int b = (int)(i / ((long long)C * oDHW));
  int z = m / (oH * oW); int r = m % (oH * oW); int y = r / oW; int xx = r % oW;
  const half_t* p = in + (size_t)b * D * H * W * C;
  float mx = -3.4e38f;
  #pragma unroll
  for (int dz = 0; dz < 2; dz++)
    #pragma unroll
    for (int dy = 0; dy < 2; dy++)
      #pragma unroll
      for (int dx = 0; dx < 2; dx++) {
        int pos = ((2 * z + dz) * H + (2 * y + dy)) * W + (2 * xx + dx);
        mx = fmaxf(mx, (float)p[(size_t)pos * C + c]);
      }
  out[i] = (half_t)mx;
}

// cat[(b*DHW+m)*2C + c] = src[(b*DHW+m)*C + c]   (first C channels of concat)
__global__ void copy_cat(const half_t* __restrict__ src, half_t* __restrict__ dst,
                         long long BM, int C) {
  long long i = (long long)blockIdx.x * blockDim.x + threadIdx.x;
  if (i >= BM * C) return;
  long long bm = i / C; int c = (int)(i % C);
  dst[bm * 2 * C + c] = src[i];
}

// out18 NDHWC (b,pos,c) == tok1 (b,t,c): straight f16->f32 convert
__global__ void tok1_kernel(const half_t* __restrict__ o18, float* __restrict__ tok, int B) {
  int i = blockIdx.x * blockDim.x + threadIdx.x;
  if (i >= B * 150 * 256) return;
  tok[i] = (float)o18[i];
}

// ---------------------------------------------------------------------------
// Head kernels (tiny, f32)
// ---------------------------------------------------------------------------
__global__ void linear_kernel(const float* __restrict__ X, const float* __restrict__ Wt,
                              const float* __restrict__ bias, float* __restrict__ Y,
                              int rows, int inF, int outF, int act) {
  int i = threadIdx.x + blockIdx.x * blockDim.x;
  if (i >= rows * outF) return;
  int o = i % outF, r = i / outF;
  const float* xp = X + (size_t)r * inF;
  const float* wp = Wt + (size_t)o * inF;
  float acc = bias ? bias[o] : 0.f;
  for (int k = 0; k < inF; k++) acc += xp[k] * wp[k];
  if (act == 1) acc = fmaxf(acc, 0.f);
  else if (act == 2) acc = 1.f / (1.f + __expf(-acc));
  Y[i] = acc;
}

__global__ void cls_kernel(const float* __restrict__ tok, const float* __restrict__ alpha,
                           const float* __restrict__ fc1w, const float* __restrict__ fc1b,
                           float* __restrict__ outCls) {
  __shared__ float s[256];
  int b = blockIdx.x, c = threadIdx.x;
  float acc = 0.f;
  for (int t = 0; t < 150; t++)
    acc += alpha[b * 150 + t] * tok[((size_t)b * 150 + t) * 256 + c];
  s[c] = acc * (1.f / 150.f);
  __syncthreads();
  if (c < 2) {
    float r = fc1b[c];
    for (int k = 0; k < 256; k++) r += s[k] * fc1w[c * 256 + k];
    outCls[b * 2 + c] = r;
  }
}

// out14 is NDHWC: o14[(b*1200 + pos)*128 + c]
__global__ void p2_kernel(const half_t* __restrict__ o14, const float* __restrict__ lw,
                          const float* __restrict__ lb, float* __restrict__ p2, int B) {
  int i = blockIdx.x * blockDim.x + threadIdx.x;
  if (i >= B * 150 * 256) return;
  int j = i & 1; int c = (i >> 1) & 127; int t = (i >> 8) % 150; int b = i / (150 * 256);
  int pz = t / 30; int rr = t % 30; int py = rr / 5; int px = rr % 5;
  const half_t* p = o14 + (size_t)b * 1200 * 128;
  float acc = lb[j];
  #pragma unroll
  for (int q = 0; q < 8; q++) {
    int dz = (q >> 2) & 1, dy = (q >> 1) & 1, dx = q & 1;
    int pos = (2 * pz + dz) * 120 + (2 * py + dy) * 10 + (2 * px + dx);
    acc += (float)p[(size_t)pos * 128 + c] * lw[j * 8 + q];
  }
  p2[((size_t)b * 150 + t) * 256 + c * 2 + j] = acc;
}

__global__ void topk_kernel(const float* __restrict__ alpha, int* __restrict__ idx, int B) {
  int b = threadIdx.x + blockIdx.x * blockDim.x;
  if (b >= B) return;
  unsigned long long used[3] = {0, 0, 0};
  for (int n = 0; n < 10; n++) {
    float best = -3.4e38f; int bi = 0;
    for (int t = 0; t < 150; t++) {
      if ((used[t >> 6] >> (t & 63)) & 1ull) continue;
      float v = alpha[b * 150 + t];
      if (v > best) { best = v; bi = t; }
    }
    used[bi >> 6] |= 1ull << (bi & 63);
    idx[b * 10 + n] = bi;
  }
}

__global__ void gather_kernel(const float* __restrict__ tok, const float* __restrict__ p2,
                              const int* __restrict__ idx, float* __restrict__ g1,
                              float* __restrict__ g2, int B) {
  int i = blockIdx.x * blockDim.x + threadIdx.x;
  if (i >= B * 10 * 256) return;
  int c = i & 255; int n = (i >> 8) % 10; int b = i / 2560;
  int t = idx[b * 10 + n];
  g1[i] = tok[((size_t)b * 150 + t) * 256 + c];
  g2[i] = p2[((size_t)b * 150 + t) * 256 + c];
}

// mode 0: raw cos matrix; 1: (cos>=0.4)+I; 2: (cos>0)+I
__global__ void cosadj_kernel(const float* __restrict__ g, float* __restrict__ out,
                              int Fdim, int mode, float eps) {
  __shared__ float nrm[10];
  int b = blockIdx.x, tid = threadIdx.x;
  if (tid < 10) {
    const float* p = g + ((size_t)b * 10 + tid) * Fdim;
    float s = 0.f;
    for (int k = 0; k < Fdim; k++) s += p[k] * p[k];
    nrm[tid] = fmaxf(sqrtf(s), eps);
  }
  __syncthreads();
  if (tid < 100) {
    int ii = tid / 10, jj = tid % 10;
    const float* pi = g + ((size_t)b * 10 + ii) * Fdim;
    const float* pj = g + ((size_t)b * 10 + jj) * Fdim;
    float d = 0.f;
    for (int k = 0; k < Fdim; k++) d += pi[k] * pj[k];
    float c = d / (nrm[ii] * nrm[jj]);
    float v;
    if (mode == 0) v = c;
    else if (mode == 1) v = (c >= 0.4f ? 1.f : 0.f) + (ii == jj ? 1.f : 0.f);
    else v = (c > 0.f ? 1.f : 0.f) + (ii == jj ? 1.f : 0.f);
    out[b * 100 + tid] = v;
  }
}

__global__ void gcn_agg_kernel(const float* __restrict__ Ah, const float* __restrict__ XW,
                               const float* __restrict__ bias, float* __restrict__ go, int B) {
  int i = blockIdx.x * blockDim.x + threadIdx.x;
  if (i >= B * 2560) return;
  int f = i & 255; int n = (i >> 8) % 10; int b = i / 2560;
  const float* A = Ah + b * 100;
  float di = 0.f;
  for (int j = 0; j < 10; j++) di += A[n * 10 + j];
  di = rsqrtf(di);
  float acc = bias[f];
  for (int j = 0; j < 10; j++) {
    float a = A[n * 10 + j];
    if (a != 0.f) {
      float dj = 0.f;
      for (int q = 0; q < 10; q++) dj += A[j * 10 + q];
      acc += di * a * rsqrtf(dj) * XW[((size_t)b * 10 + j) * 256 + f];
    }
  }
  go[i] = fmaxf(acc, 0.f);
}

__global__ void fus_kernel(const float* __restrict__ go1, const float* __restrict__ go2,
                           float* __restrict__ fus, int B) {
  int i = blockIdx.x * blockDim.x + threadIdx.x;
  if (i >= B * 2560) return;
  int f = i & 255; int n = (i >> 8) % 10; int b = i / 2560;
  fus[(((size_t)b * 10 + n) * 2 + 0) * 256 + f] = go1[i];
  fus[(((size_t)b * 10 + n) * 2 + 1) * 256 + f] = go2[i];
}

__global__ void sa_kernel(const float* __restrict__ q, const float* __restrict__ k,
                          const float* __restrict__ v, float* __restrict__ gcat, int B) {
  int i = threadIdx.x + blockIdx.x * blockDim.x;
  if (i >= B * 20) return;
  int ii = i & 1; int n = (i >> 1) % 10; int b = i / 20;
  const float* qi = q + (((size_t)b * 10 + n) * 2 + ii) * 256;
  const float* kb = k + (((size_t)b * 10 + n) * 2) * 256;
  const float* vb = v + (((size_t)b * 10 + n) * 2) * 256;
  float s0 = 0.f, s1 = 0.f;
  for (int f = 0; f < 256; f++) { s0 += qi[f] * kb[f]; s1 += qi[f] * kb[256 + f]; }
  s0 *= (1.f / 16.f); s1 *= (1.f / 16.f);
  float m = fmaxf(s0, s1);
  float e0 = __expf(s0 - m), e1 = __expf(s1 - m);
  float inv = 1.f / (e0 + e1);
  e0 *= inv; e1 *= inv;
  float* o = gcat + ((size_t)b * 10 + n) * 512 + ii * 256;
  for (int f = 0; f < 256; f++) o[f] = e0 * vb[f] + e1 * vb[256 + f];
}

__global__ void agnn_w_kernel(const float* __restrict__ Ah, const float* __restrict__ cm,
                              const float* __restrict__ beta, float* __restrict__ Wn, int B) {
  int i = threadIdx.x + blockIdx.x * blockDim.x;
  if (i >= B * 10) return;
  int n = i % 10; int b = i / 10;
  float bv = beta[0];
  float w[10]; float s = 0.f;
  for (int j = 0; j < 10; j++) {
    float ww = Ah[b * 100 + n * 10 + j] * __expf(bv * cm[b * 100 + n * 10 + j]);
    w[j] = ww; s += ww;
  }
  float inv = 1.f / s;
  for (int j = 0; j < 10; j++) Wn[b * 100 + n * 10 + j] = w[j] * inv;
}

__global__ void agnn_prop_kernel(const float* __restrict__ Wn, const float* __restrict__ h,
                                 float* __restrict__ gsum, int B) {
  int i = threadIdx.x + blockIdx.x * blockDim.x;
  if (i >= B * 256) return;
  int f = i & 255; int b = i >> 8;
  float acc = 0.f;
  for (int n = 0; n < 10; n++) {
    float row = 0.f;
    for (int j = 0; j < 10; j++)
      row += Wn[b * 100 + n * 10 + j] * h[((size_t)b * 10 + j) * 256 + f];
    acc += row;
  }
  gsum[i] += acc * 0.1f;
}

__global__ void zero_kernel(float* __restrict__ p, int n) {
  int i = threadIdx.x + blockIdx.x * blockDim.x;
  if (i < n) p[i] = 0.f;
}

__global__ void copy_f32(const float* __restrict__ s, float* __restrict__ d, int n) {
  int i = threadIdx.x + blockIdx.x * blockDim.x;
  if (i < n) d[i] = s[i];
}

// ---------------------------------------------------------------------------
// Host-side orchestration
// ---------------------------------------------------------------------------
struct CSpec { int ic, oc, ks; };
static const CSpec g_cs[11] = {{1,16,2},{16,16,3},{16,16,3},{32,32,3},{32,32,3},
                               {64,64,3},{64,64,3},{128,128,1},{128,128,1},
                               {256,256,1},{256,256,1}};
static const size_t g_whoff[10] = {0,7168,14336,43008,71680,182272,292864,309248,325632,391168};

static inline int nb(long long n, int t) { return (int)((n + t - 1) / t); }

extern "C" void kernel_launch(void* const* d_in, const int* in_sizes, int n_in,
                              void* d_out, int out_size, void* d_ws, size_t ws_size,
                              hipStream_t stream) {
  (void)in_sizes; (void)n_in; (void)out_size; (void)ws_size;
  const int B = 2;
  char* ws = (char*)d_ws;
  float* out = (float*)d_out;

  auto Fp = [&](int i) { return (const float*)d_in[i]; };
  auto HB = [&](size_t off) { return (half_t*)(ws + off); };
  auto FB = [&](size_t off) { return (float*)(ws + off); };
  half_t* WF = HB(OFF_W);
  auto SB = [&](int i) { return (float*)(ws + OFF_SB + (size_t)i * 4096); };

  // ---- fold BN / swizzle weights ----
  for (int i = 0; i < 11; i++)
    prep_sb<<<1, 256, 0, stream>>>(Fp(3 + 6 * i), Fp(4 + 6 * i), Fp(5 + 6 * i),
                                   Fp(6 + 6 * i), Fp(2 + 6 * i), SB(i), g_cs[i].oc);
  for (int i = 1; i < 11; i++) {
    int ks3 = g_cs[i].ks * g_cs[i].ks * g_cs[i].ks;
    int K = g_cs[i].ic * ks3;
    int Kp = (K + 31) & ~31;
    int tot = g_cs[i].oc * Kp;
    prep_wfrag<<<nb(tot, 256), 256, 0, stream>>>(Fp(1 + 6 * i), WF + g_whoff[i - 1],
                                                 g_cs[i].oc, g_cs[i].ic, ks3);
  }

  // ---- stem ----
  stem_kernel<<<nb((long long)B * 16 * 614400, 256), 256, 0, stream>>>(
      Fp(0), Fp(1), SB(0), HB(OFF_H), B);

  auto conv = [&](int wi, const half_t* in, const half_t* aIn, half_t* o, half_t* o2,
                  int D, int H, int W, int pad, int mode, int outC, int outCOff) {
    int IC = g_cs[wi].ic, OC = g_cs[wi].oc, ks = g_cs[wi].ks;
    int DHW = D * H * W;
    dim3 g((DHW + 63) / 64, OC / 16, B);
    const half_t* wf = WF + g_whoff[wi - 1];
    if (ks == 3 && IC == 16)
      conv_wmma_kernel<3, 16><<<g, 128, 0, stream>>>(in, wf, SB(wi), aIn, o, o2,
                                                     OC, D, H, W, pad, mode, outC, outCOff);
    else if (ks == 3 && IC == 32)
      conv_wmma_kernel<3, 32><<<g, 128, 0, stream>>>(in, wf, SB(wi), aIn, o, o2,
                                                     OC, D, H, W, pad, mode, outC, outCOff);
    else if (ks == 3 && IC == 64)
      conv_wmma_kernel<3, 64><<<g, 128, 0, stream>>>(in, wf, SB(wi), aIn, o, o2,
                                                     OC, D, H, W, pad, mode, outC, outCOff);
    else if (ks == 1 && IC == 128)
      conv_wmma_kernel<1, 128><<<g, 128, 0, stream>>>(in, wf, SB(wi), aIn, o, o2,
                                                      OC, D, H, W, pad, mode, outC, outCOff);
    else
      conv_wmma_kernel<1, 256><<<g, 128, 0, stream>>>(in, wf, SB(wi), aIn, o, o2,
                                                      OC, D, H, W, pad, mode, outC, outCOff);
  };

  // ---- residual stages: a=relu(bn(conv)); cat=[h, relu(a+bn(conv(a)))]; pool ----
  struct SDim { int C, D, H, W, pad; };
  const SDim sd[4] = {{16, 80, 96, 80, 1}, {32, 40, 48, 40, 1},
                      {64, 20, 24, 20, 1}, {128, 10, 12, 10, 0}};
  for (int s = 0; s < 4; s++) {
    int C = sd[s].C, D = sd[s].D, H = sd[s].H, W = sd[s].W, pad = sd[s].pad;
    long long DHW = (long long)D * H * W;
    int wa = 1 + 2 * s, wb = 2 + 2 * s;
    conv(wa, HB(OFF_H), nullptr, HB(OFF_A), nullptr, D, H, W, pad, 1, C, 0);
    conv(wb, HB(OFF_A), HB(OFF_A), HB(OFF_CAT),
         (s == 3) ? HB(OFF_B2) : nullptr, D, H, W, pad, 2, 2 * C, C);
    copy_cat<<<nb((long long)B * DHW * C, 256), 256, 0, stream>>>(
        HB(OFF_H), HB(OFF_CAT), (long long)B * DHW, C);
    pool_kernel<<<nb((long long)B * 2 * C * (DHW / 8), 256), 256, 0, stream>>>(
        HB(OFF_CAT), HB(OFF_H), B, 2 * C, D, H, W);
  }

  // ---- 1x1 convs 9/10 on (B,150,256) NDHWC ----
  conv(9,  HB(OFF_H), nullptr, HB(OFF_A),   nullptr, 5, 6, 5, 0, 1, 256, 0);
  conv(10, HB(OFF_A), nullptr, HB(OFF_CAT), nullptr, 5, 6, 5, 0, 1, 256, 0); // out18

  // ---- head ----
  float* TOK = FB(OFF_TOK); float* P2 = FB(OFF_P2); float* HH = FB(OFF_HH);
  float* AL = FB(OFF_AL);   int* IDX = (int*)(ws + OFF_IDX);
  float* G1 = FB(OFF_G1);   float* G2 = FB(OFF_G2);
  float* AH1 = FB(OFF_AH1); float* AH2 = FB(OFF_AH2);
  float* XW = FB(OFF_XW);   float* GO1 = FB(OFF_GO1); float* GO2 = FB(OFF_GO2);
  float* FUS = FB(OFF_FUS); float* Q = FB(OFF_Q); float* Kb = FB(OFF_K); float* V = FB(OFF_V);
  float* GC = FB(OFF_GC);   float* HA = FB(OFF_HA); float* CM = FB(OFF_CM);
  float* WN = FB(OFF_WN);   float* GS = FB(OFF_GS);

  tok1_kernel<<<nb(B * 150 * 256, 256), 256, 0, stream>>>(HB(OFF_CAT), TOK, B);
  // f = tok1 @ fc1^T + b
  linear_kernel<<<nb(300 * 2, 256), 256, 0, stream>>>(TOK, Fp(71), Fp(72), out + 4,
                                                      300, 256, 2, 0);
  // attention MLP -> alpha
  linear_kernel<<<nb(300 * 64, 256), 256, 0, stream>>>(TOK, Fp(67), Fp(68), HH,
                                                       300, 256, 64, 1);
  linear_kernel<<<nb(300, 256), 256, 0, stream>>>(HH, Fp(69), Fp(70), AL, 300, 64, 1, 2);
  copy_f32<<<nb(300, 256), 256, 0, stream>>>(AL, out + 604, 300);
  cls_kernel<<<B, 256, 0, stream>>>(TOK, AL, Fp(71), Fp(72), out);
  // unfold(out14)+lin1 -> p2
  p2_kernel<<<nb(B * 150 * 256, 256), 256, 0, stream>>>(HB(OFF_B2), Fp(73), Fp(74), P2, B);
  topk_kernel<<<1, 32, 0, stream>>>(AL, IDX, B);
  gather_kernel<<<nb(B * 2560, 256), 256, 0, stream>>>(TOK, P2, IDX, G1, G2, B);
  cosadj_kernel<<<B, 128, 0, stream>>>(G1, AH1, 256, 1, 1e-8f);
  cosadj_kernel<<<B, 128, 0, stream>>>(G2, AH2, 256, 2, 1e-8f);
  // GCNs
  linear_kernel<<<nb(20 * 256, 256), 256, 0, stream>>>(G1, Fp(75), nullptr, XW, 20, 256, 256, 0);
  gcn_agg_kernel<<<nb(B * 2560, 256), 256, 0, stream>>>(AH1, XW, Fp(76), GO1, B);
  linear_kernel<<<nb(20 * 256, 256), 256, 0, stream>>>(G2, Fp(77), nullptr, XW, 20, 256, 256, 0);
  gcn_agg_kernel<<<nb(B * 2560, 256), 256, 0, stream>>>(AH2, XW, Fp(78), GO2, B);
  // self-attention over the 2 fused tokens
  fus_kernel<<<nb(B * 2560, 256), 256, 0, stream>>>(GO1, GO2, FUS, B);
  linear_kernel<<<nb(40 * 256, 256), 256, 0, stream>>>(FUS, Fp(79), nullptr, Q, 40, 256, 256, 0);
  linear_kernel<<<nb(40 * 256, 256), 256, 0, stream>>>(FUS, Fp(80), nullptr, Kb, 40, 256, 256, 0);
  linear_kernel<<<nb(40 * 256, 256), 256, 0, stream>>>(FUS, Fp(81), nullptr, V, 40, 256, 256, 0);
  sa_kernel<<<nb(B * 20, 64), 64, 0, stream>>>(Q, Kb, V, GC, B);
  // AGNN x2 accumulated (mean over nodes folded in as *0.1)
  zero_kernel<<<nb(B * 256, 256), 256, 0, stream>>>(GS, B * 256);
  for (int a = 0; a < 2; a++) {
    const float* Wg = Fp(82 + 3 * a); const float* bg = Fp(83 + 3 * a);
    const float* bt = Fp(84 + 3 * a);
    const float* Ah = (a == 0) ? AH1 : AH2;
    linear_kernel<<<nb(20 * 256, 256), 256, 0, stream>>>(GC, Wg, bg, HA, 20, 512, 256, 1);
    cosadj_kernel<<<B, 128, 0, stream>>>(HA, CM, 256, 0, 1e-12f);
    agnn_w_kernel<<<1, 32, 0, stream>>>(Ah, CM, bt, WN, B);
    agnn_prop_kernel<<<nb(B * 256, 256), 256, 0, stream>>>(WN, HA, GS, B);
  }
  // final = (g1o+g2o) @ fc2^T + b
  linear_kernel<<<1, 256, 0, stream>>>(GS, Fp(88), Fp(89), out + 904, B, 256, 2, 0);
}
